// UFOAtt_24094766530810
// MI455X (gfx1250) — matrix operand order repaired and checked by reading, not compiled, not run on hardware
//
#include <hip/hip_runtime.h>
#include <hip/hip_bf16.h>

// ---------------------------------------------------------------------------
// UFO linear attention, fused bf16-WMMA pipeline for gfx1250 (MI455X).
// Memory-bound (~270 MB min traffic @ 23.3 TB/s). All GEMMs use
// v_wmma_f32_16x16x32_bf16. The output projection Wo is folded into the
// per-(b,h) normalized kv matrices (M2[b] = stack_h(kv_n @ Wo_h^T)), so the
// [B,H,N,DV] intermediate never touches HBM.
//
// Data-movement strategy (this round):
//  * All WMMA "B" operands stored pair-interleaved [K/2][N][2] in bf16 so a
//    B fragment is 8 lane-coalesced dword loads (one (K,K+1) pair per VGPR).
//  * k/v projections are pair-interleaved over the token dim at store time,
//    making the K^T V reduction's A and B fragments dword loads too.
//  * Projection activations (f32) staged through LDS (coalesced float2
//    loads, bf16 in LDS, fragments via 2x ds_load_b128 per lane).
//  * Per-wave tiles sized so accumulators never spill (<=8 v8f per wave in
//    the big kernels, 4 in the hot ones).
// ---------------------------------------------------------------------------

typedef __bf16 bhalf;
typedef __attribute__((ext_vector_type(16))) __bf16        v16bf;
typedef __attribute__((ext_vector_type(8)))  float         v8f;
typedef __attribute__((ext_vector_type(8)))  unsigned int  v8u;

#define NB 8
#define NN 4096
#define NC 512
#define NH 8
#define ND 64   // DK == DV == 64

static __device__ __forceinline__ v8f wmma_bf16(v16bf a, v16bf b, v8f c) {
  return __builtin_amdgcn_wmma_f32_16x16x32_bf16(false, a, false, b, (short)0, c,
                                                 false, false);
}

static __device__ __forceinline__ v16bf u2bf(v8u u) {
  union { v8u u; v16bf v; } x; x.u = u; return x.v;
}

// --- A fragment from row-major bf16 (pairs along K are contiguous):
// lane L<16: row M=L, K chunks {half*8.., 16+half*8..} -> two 16B loads.
static __device__ __forceinline__ v16bf frag_a_row(const bhalf* base, int lda,
                                                   int lane) {
  const int m = lane & 15, half = (lane >> 4) & 1;
  const bhalf* p = base + (size_t)m * lda + half * 8;
  union { uint4 q[2]; v16bf v; } u;
  u.q[0] = *(const uint4*)(p);
  u.q[1] = *(const uint4*)(p + 16);
  return u.v;
}

// --- A fragment from a 16x32 bf16 tile in LDS (row stride 32).
static __device__ __forceinline__ v16bf frag_a_lds(const bhalf* base, int lane) {
  const int m = lane & 15, half = (lane >> 4) & 1;
  const bhalf* p = base + m * 32 + half * 8;
  union { uint4 q[2]; v16bf v; } u;
  u.q[0] = *(const uint4*)(p);
  u.q[1] = *(const uint4*)(p + 16);
  return u.v;
}

// --- A fragment from row-major f32 (m2 kernel): convert float2 pairs.
static __device__ __forceinline__ v16bf frag_a_f32row(const float* base, int lda,
                                                      int lane) {
  const int m = lane & 15, half = (lane >> 4) & 1;
  const float* p = base + (size_t)m * lda + half * 8;
  v16bf a;
#pragma unroll
  for (int v = 0; v < 8; ++v) {
    const int kb = ((v & 4) ? 16 : 0) + (v & 3) * 2;
    const float2 f = *(const float2*)(p + kb);
    a[2 * v]     = (bhalf)f.x;
    a[2 * v + 1] = (bhalf)f.y;
  }
  return a;
}

// --- B fragment from pair-interleaved [K/2][N][2] bf16 viewed as u32[K/2][N].
// base must point at (pair k0/2, col n0). One dword = (B[k][n], B[k+1][n]).
static __device__ __forceinline__ v16bf frag_b_pairs(const unsigned* base,
                                                     int ldn, int lane) {
  const int n = lane & 15, half = (lane >> 4) & 1;
  const unsigned* p = base + (size_t)(half * 4) * ldn + n;
  v8u u;
#pragma unroll
  for (int v = 0; v < 8; ++v) u[v] = p[(size_t)(((v & 4) ? 8 : 0) + (v & 3)) * ldn];
  return u2bf(u);
}

// --- A fragment (transposed view, K = token dim) from pair-interleaved
// u32[K/2][N]; base at (pair n0/2, channel ch0). a[2v] pairs map to kp below.
static __device__ __forceinline__ v16bf frag_a_pairs(const unsigned* base,
                                                     int ldn, int lane) {
  const int m = lane & 15, half = (lane >> 4) & 1;
  v8u u;
#pragma unroll
  for (int v = 0; v < 8; ++v) {
    const int kp = ((v & 4) ? 8 : 0) + half * 4 + (v & 3);  // = kb/2
    u[v] = base[(size_t)kp * ldn + m];
  }
  return u2bf(u);
}

// ---------------------------------------------------------------------------
// K1: f32 [512 out][512 in] weight -> bf16 pair-interleaved [in/2][out][2]
// (B-operand layout for all four weight matrices; same index map for Wo).
// ---------------------------------------------------------------------------
__global__ void wprep_kernel(const float* __restrict__ Wq,
                             const float* __restrict__ Wk,
                             const float* __restrict__ Wv,
                             const float* __restrict__ Wo,
                             bhalf* __restrict__ WPq, bhalf* __restrict__ WPk,
                             bhalf* __restrict__ WPv, bhalf* __restrict__ WoP) {
  const int idx = blockIdx.x * 256 + threadIdx.x;   // 0 .. 512*512-1
  const int row = idx >> 9;     // out-dim (WMMA N)
  const int col = idx & 511;    // in-dim  (WMMA K)
  const float* src;
  bhalf* dst;
  switch (blockIdx.y) {
    case 0:  src = Wq; dst = WPq; break;
    case 1:  src = Wk; dst = WPk; break;
    case 2:  src = Wv; dst = WPv; break;
    default: src = Wo; dst = WoP; break;
  }
  dst[((size_t)(col >> 1) * NC + row) * 2 + (col & 1)] = (bhalf)src[idx];
}

// ---------------------------------------------------------------------------
// K2: projection GEMM  act[32768 x 512](f32) @ W^T via pair-interleaved bf16.
// Block = 16 rows x 512 cols; 8 waves x (16x64). A staged through LDS.
// q stored f32 row-major (norms need f32 rows); k/v stored bf16
// pair-interleaved over the token dim for the K^T V kernel.
// ---------------------------------------------------------------------------
template <bool OUT_F32>
__global__ void proj_kernel(const float* __restrict__ act,
                            const bhalf* __restrict__ WP,
                            const float* __restrict__ bias,
                            float* __restrict__ outF,
                            bhalf* __restrict__ outP) {
  __shared__ __align__(16) bhalf als[2][16 * 32];
  const int tid = threadIdx.x, lane = tid & 31, wave = tid >> 5;
  const int row0 = blockIdx.x * 16;
  const int col0 = wave * 64;
  const unsigned* WPu = (const unsigned*)WP;

  const v8f zero = {0.f, 0.f, 0.f, 0.f, 0.f, 0.f, 0.f, 0.f};
  v8f acc[4] = {zero, zero, zero, zero};

  const int sr = tid >> 4, sc = (tid & 15) * 2;   // staging: 16 rows x 16 f2
  for (int ks = 0; ks < 16; ++ks) {
    const int kk = ks * 32;
    bhalf* buf = als[ks & 1];
    const float2 f =
        *(const float2*)(act + (size_t)(row0 + sr) * NC + kk + sc);
    buf[sr * 32 + sc]     = (bhalf)f.x;
    buf[sr * 32 + sc + 1] = (bhalf)f.y;
    __syncthreads();

    const v16bf a = frag_a_lds(buf, lane);
#pragma unroll
    for (int t = 0; t < 4; ++t) {
      const v16bf b = frag_b_pairs(
          WPu + (size_t)(kk >> 1) * NC + col0 + t * 16, NC, lane);
      acc[t] = wmma_bf16(a, b, acc[t]);
    }
  }

  const int half = lane >> 4, cn = lane & 15;
#pragma unroll
  for (int t = 0; t < 4; ++t) {
#pragma unroll
    for (int j = 0; j < 8; ++j) {
      const int r = row0 + (half << 3) + j;   // token row
      const int c = col0 + t * 16 + cn;       // channel
      const float v = acc[t][j] + bias[c];
      if (OUT_F32) {
        outF[(size_t)r * NC + c] = v;
      } else {
        // pair-interleave over token dim: [(r/2)][c][r&1]
        outP[((size_t)(r >> 1) * NC + c) * 2 + (r & 1)] = (bhalf)v;
      }
    }
  }
}

// ---------------------------------------------------------------------------
// K3: per-(b,h)  kv = K^T V (64x64, reduce over N=4096) + row L2-norm*gamma.
// 8 waves: wave = (qr, qc, kslice); each wave owns a 32x32 quadrant over a
// 2048-token slice -> 4 accumulators, no spills. All loads are dword loads
// from the pair-interleaved k/v. LDS f32 reduction across the 2 K-slices.
// ---------------------------------------------------------------------------
__global__ void kv_kernel(const bhalf* __restrict__ PK,
                          const bhalf* __restrict__ PV,
                          const float* __restrict__ gamma,
                          float* __restrict__ kvn) {
  __shared__ float red[ND * ND];
  const int tid = threadIdx.x, lane = tid & 31, wave = tid >> 5;
  const int bh = blockIdx.x, b = bh >> 3, h = bh & 7;
  const int qr = wave & 1, qc = (wave >> 1) & 1, slice = wave >> 2;

  for (int i = tid; i < ND * ND; i += 256) red[i] = 0.f;
  __syncthreads();

  const unsigned* PKu = (const unsigned*)PK;
  const unsigned* PVu = (const unsigned*)PV;
  const int chK = h * ND + qr * 32;
  const int chV = h * ND + qc * 32;

  const v8f zero = {0.f, 0.f, 0.f, 0.f, 0.f, 0.f, 0.f, 0.f};
  v8f acc[4] = {zero, zero, zero, zero};

  const int nEnd = slice * 2048 + 2048;
  for (int n0 = slice * 2048; n0 < nEnd; n0 += 32) {
    const size_t pbase = ((size_t)(b * NN + n0) >> 1) * NC;
    const unsigned* pk = PKu + pbase + chK;
    const unsigned* pv = PVu + pbase + chV;
    const v16bf a0 = frag_a_pairs(pk, NC, lane);
    const v16bf a1 = frag_a_pairs(pk + 16, NC, lane);
    const v16bf b0 = frag_b_pairs(pv, NC, lane);
    const v16bf b1 = frag_b_pairs(pv + 16, NC, lane);
    acc[0] = wmma_bf16(a0, b0, acc[0]);
    acc[1] = wmma_bf16(a0, b1, acc[1]);
    acc[2] = wmma_bf16(a1, b0, acc[2]);
    acc[3] = wmma_bf16(a1, b1, acc[3]);
  }

  const int half = lane >> 4, cn = lane & 15;
#pragma unroll
  for (int rt = 0; rt < 2; ++rt)
#pragma unroll
    for (int ct = 0; ct < 2; ++ct)
#pragma unroll
      for (int j = 0; j < 8; ++j)
        atomicAdd(&red[(qr * 32 + rt * 16 + (half << 3) + j) * ND +
                       qc * 32 + ct * 16 + cn],
                  acc[rt * 2 + ct][j]);
  __syncthreads();

  if (tid < ND) {
    float ss = 0.f;
#pragma unroll 8
    for (int c = 0; c < ND; ++c) { const float x = red[tid * ND + c]; ss += x * x; }
    const float scale = gamma[h] / sqrtf(ss);
    float* dst = kvn + ((size_t)bh * ND + tid) * ND;
#pragma unroll 8
    for (int c = 0; c < ND; ++c) dst[c] = red[tid * ND + c] * scale;
  }
}

// ---------------------------------------------------------------------------
// K3b: normalize q rows (64 values per (b,n,h)) -> bf16 q_n. One wave per row.
// ---------------------------------------------------------------------------
__global__ void qnorm_kernel(const float* __restrict__ qf,
                             const float* __restrict__ gamma,
                             bhalf* __restrict__ qn) {
  const int lane = threadIdx.x & 31, wave = threadIdx.x >> 5;
  const size_t idx = (size_t)blockIdx.x * 8 + wave;  // flat (b,n,h) row id
  const float2 x = *(const float2*)(qf + idx * ND + 2 * lane);
  float ss = x.x * x.x + x.y * x.y;
#pragma unroll
  for (int m = 16; m >= 1; m >>= 1) ss += __shfl_xor(ss, m, 32);
  const float scale = gamma[(int)(idx & 7)] / sqrtf(ss);
  qn[idx * ND + 2 * lane]     = (bhalf)(x.x * scale);
  qn[idx * ND + 2 * lane + 1] = (bhalf)(x.y * scale);
}

// ---------------------------------------------------------------------------
// K4: per-(b,h)  M2 = kv_n (64x64) @ Wo_h^T (64x512), written bf16
// pair-interleaved over its K dim (= rows h*64+dk) for the final GEMM.
// grid (B*H, 2 col-halves); 8 waves = (row-half, 4 col-strips): 8 acc each.
// ---------------------------------------------------------------------------
__global__ void m2_kernel(const float* __restrict__ kvn,
                          const bhalf* __restrict__ WoP,
                          bhalf* __restrict__ M2P) {
  const int lane = threadIdx.x & 31, wave = threadIdx.x >> 5;
  const int bh = blockIdx.x, b = bh >> 3, h = bh & 7;
  const int rh = wave & 1;                       // 32-row half
  const int col0 = blockIdx.y * 256 + (wave >> 1) * 64;
  const float* A = kvn + (size_t)bh * ND * ND;
  const unsigned* WoPu = (const unsigned*)WoP;

  const v8f zero = {0.f, 0.f, 0.f, 0.f, 0.f, 0.f, 0.f, 0.f};
  v8f acc[8];
#pragma unroll
  for (int t = 0; t < 8; ++t) acc[t] = zero;

#pragma unroll
  for (int kk = 0; kk < ND; kk += 32) {
    v16bf afr[2];
#pragma unroll
    for (int rt = 0; rt < 2; ++rt)
      afr[rt] = frag_a_f32row(A + (size_t)(rh * 32 + rt * 16) * ND + kk, ND, lane);
#pragma unroll
    for (int ct = 0; ct < 4; ++ct) {
      // K dim of WoP is Wo's in-dim j = h*64 + kk + ...
      const v16bf bfr = frag_b_pairs(
          WoPu + (size_t)((h * ND + kk) >> 1) * NC + col0 + ct * 16, NC, lane);
#pragma unroll
      for (int rt = 0; rt < 2; ++rt)
        acc[rt * 4 + ct] = wmma_bf16(afr[rt], bfr, acc[rt * 4 + ct]);
    }
  }

  const int half = lane >> 4, cn = lane & 15;
  bhalf* dst = M2P + (size_t)b * NC * NC;
#pragma unroll
  for (int rt = 0; rt < 2; ++rt)
#pragma unroll
    for (int ct = 0; ct < 4; ++ct)
#pragma unroll
      for (int j = 0; j < 8; ++j) {
        const int rk = h * ND + rh * 32 + rt * 16 + (half << 3) + j;  // K row
        const int c  = col0 + ct * 16 + cn;
        dst[((size_t)(rk >> 1) * NC + c) * 2 + (rk & 1)] =
            (bhalf)acc[rt * 4 + ct][j];
      }
}

// ---------------------------------------------------------------------------
// K5: per-batch  out[b] = q_n[b] (4096x512) @ M2[b] (512x512) + bo -> f32.
// A fragments: two global_load_b128 per lane; B: dword pair loads.
// ---------------------------------------------------------------------------
__global__ void final_kernel(const bhalf* __restrict__ qn,
                             const bhalf* __restrict__ M2P,
                             const float* __restrict__ bo,
                             float* __restrict__ out) {
  const int lane = threadIdx.x & 31, wave = threadIdx.x >> 5;
  const int b = blockIdx.y;
  const int row0 = blockIdx.x * 16;
  const bhalf* A = qn + ((size_t)b * NN + row0) * NC;
  const unsigned* Bm = (const unsigned*)M2P + (size_t)b * (NC / 2) * NC;
  const int col0 = wave * 64;

  const v8f zero = {0.f, 0.f, 0.f, 0.f, 0.f, 0.f, 0.f, 0.f};
  v8f acc[4] = {zero, zero, zero, zero};

  for (int kk = 0; kk < NC; kk += 32) {
    const v16bf a = frag_a_row(A + kk, NC, lane);
#pragma unroll
    for (int t = 0; t < 4; ++t) {
      const v16bf bfr = frag_b_pairs(
          Bm + (size_t)(kk >> 1) * NC + col0 + t * 16, NC, lane);
      acc[t] = wmma_bf16(a, bfr, acc[t]);
    }
  }

  const int half = lane >> 4, cn = lane & 15;
#pragma unroll
  for (int t = 0; t < 4; ++t)
#pragma unroll
    for (int j = 0; j < 8; ++j) {
      const int c = col0 + t * 16 + cn;
      out[((size_t)b * NN + row0 + (half << 3) + j) * NC + c] =
          acc[t][j] + bo[c];
    }
}

// ---------------------------------------------------------------------------
extern "C" void kernel_launch(void* const* d_in, const int* in_sizes, int n_in,
                              void* d_out, int out_size, void* d_ws,
                              size_t ws_size, hipStream_t stream) {
  (void)in_sizes; (void)n_in; (void)out_size; (void)ws_size;
  const float* queries = (const float*)d_in[0];
  const float* keys    = (const float*)d_in[1];
  const float* values  = (const float*)d_in[2];
  const float* Wq = (const float*)d_in[3];
  const float* bq = (const float*)d_in[4];
  const float* Wk = (const float*)d_in[5];
  const float* bk = (const float*)d_in[6];
  const float* Wv = (const float*)d_in[7];
  const float* bv = (const float*)d_in[8];
  const float* Wo = (const float*)d_in[9];
  const float* bo = (const float*)d_in[10];
  const float* gamma = (const float*)d_in[11];
  float* outp = (float*)d_out;

  char* w = (char*)d_ws;
  size_t off = 0;
  auto carve = [&](size_t bytes) -> void* {
    void* p = (void*)(w + off);
    off += (bytes + 255) & ~(size_t)255;
    return p;
  };
  bhalf* WPq = (bhalf*)carve((size_t)NC * NC * 2);
  bhalf* WPk = (bhalf*)carve((size_t)NC * NC * 2);
  bhalf* WPv = (bhalf*)carve((size_t)NC * NC * 2);
  bhalf* WoP = (bhalf*)carve((size_t)NC * NC * 2);
  float* qf32 = (float*)carve((size_t)NB * NN * NC * 4);
  bhalf* PK   = (bhalf*)carve((size_t)NB * NN * NC * 2);
  bhalf* PV   = (bhalf*)carve((size_t)NB * NN * NC * 2);
  bhalf* qn   = (bhalf*)carve((size_t)NB * NN * NC * 2);
  float* kvn  = (float*)carve((size_t)NB * NH * ND * ND * 4);
  bhalf* M2P  = (bhalf*)carve((size_t)NB * NC * NC * 2);

  // K1: weights -> pair-interleaved bf16
  wprep_kernel<<<dim3((NC * NC) / 256, 4), 256, 0, stream>>>(
      Wq, Wk, Wv, Wo, WPq, WPk, WPv, WoP);

  // K2: projections (M = B*N = 32768 rows, 16-row tiles)
  const int mblk = (NB * NN) / 16;
  proj_kernel<true ><<<dim3(mblk), 256, 0, stream>>>(queries, WPq, bq, qf32, nullptr);
  proj_kernel<false><<<dim3(mblk), 256, 0, stream>>>(keys,    WPk, bk, nullptr, PK);
  proj_kernel<false><<<dim3(mblk), 256, 0, stream>>>(values,  WPv, bv, nullptr, PV);

  // K3: kv = K^T V + row-norm*gamma
  kv_kernel<<<dim3(NB * NH), 256, 0, stream>>>(PK, PV, gamma, kvn);

  // K3b: q row norms -> bf16
  qnorm_kernel<<<dim3((NB * NN * NH) / 8), 256, 0, stream>>>(qf32, gamma, qn);

  // K4: fold Wo into kv_n -> M2 (pair-interleaved)
  m2_kernel<<<dim3(NB * NH, 2), 256, 0, stream>>>(kvn, WoP, M2P);

  // K5: out = q_n @ M2 + bo
  final_kernel<<<dim3(NN / 16, NB), 256, 0, stream>>>(qn, M2P, bo, outp);
}
